// DeepseekV32Attention_43963285242655
// MI455X (gfx1250) — matrix operand order repaired and checked by reading, not compiled, not run on hardware
//
#include <hip/hip_runtime.h>
#include <hip/hip_bf16.h>

// ---------------------------------------------------------------------------
// DeepSeek V3.2 sparse MLA attention for gfx1250 (MI455X).
// All matrix math: v_wmma_f32_16x16x32_bf16 (bf16 in, fp32 accum).
// Attention K/V/k_pe are materialized in bf16 (K pre-transposed) so the
// flash-attention tiles are staged with the Tensor Data Mover when available.
// ---------------------------------------------------------------------------

typedef __attribute__((ext_vector_type(8)))  float          v8f;
typedef __attribute__((ext_vector_type(16))) unsigned short v16u16;
typedef __attribute__((ext_vector_type(8)))  unsigned short v8u16;
typedef __attribute__((ext_vector_type(16))) __bf16         v16bf;

static constexpr int L_      = 1024;
static constexpr int D_      = 4096;
static constexpr int H_      = 32;
static constexpr int QLORA_  = 1536;
static constexpr int KVLORA_ = 512;
static constexpr int ROPED_  = 64;
static constexpr int NOPED_  = 128;
static constexpr int VD_     = 128;
static constexpr int QHD_    = 192;   // NOPE_D + ROPE_D
static constexpr int IH_     = 64;
static constexpr int IHD_    = 128;
static constexpr int TOPK_   = 256;

__device__ __forceinline__ unsigned short f2bf(float f){
  unsigned u = __float_as_uint(f);
  u += 0x7FFFu + ((u >> 16) & 1u);          // round-to-nearest-even
  return (unsigned short)(u >> 16);
}

__device__ __forceinline__ v8f zero8(){
  v8f z;
#pragma unroll
  for (int i = 0; i < 8; ++i) z[i] = 0.f;
  return z;
}

__device__ __forceinline__ v8f wmma_bf16(v16u16 a, v16u16 b, v8f c){
  return __builtin_amdgcn_wmma_f32_16x16x32_bf16(
      false, __builtin_bit_cast(v16bf, a),
      false, __builtin_bit_cast(v16bf, b),
      (short)0, c, false, false);
}

// A fragment (16x32 bf16) from LDS tile stored [m][k] row-major (stride in u16).
__device__ __forceinline__ v16u16 ld_afrag(const unsigned short* base, int stride,
                                           int m, int k0, int khalf){
  union { v16u16 v; v8u16 h[2]; } f;
  const unsigned short* p = base + m * stride + k0 + khalf * 8;
  f.h[0] = *reinterpret_cast<const v8u16*>(p);
  f.h[1] = *reinterpret_cast<const v8u16*>(p + 16);
  return f.v;
}

// B fragment (32x16 bf16): k = kbase + lane; lane%16 = N, elements walk K (table order); LDS tile stored [k][n] (stride in u16).
__device__ __forceinline__ v16u16 ld_bfrag(const unsigned short* base, int stride,
                                           int k, int n0){
  v16u16 v;
  const int ln = k & 31, kb = k - ln, hh = ln >> 4;
  const unsigned short* p = base + n0 + (ln & 15);
#pragma unroll
  for (int e = 0; e < 8; ++e){
    v[e]     = p[(kb + 8 * hh + e) * stride];
    v[8 + e] = p[(kb + 16 + 8 * hh + e) * stride];
  }
  return v;
}

// A fragment straight from a global fp32 row pointer (scale folded in).
__device__ __forceinline__ v16u16 ld_afrag_global(const float* rowp, int k0,
                                                  int khalf, float scale){
  v16u16 v;
#pragma unroll
  for (int e = 0; e < 8; ++e) v[e]     = f2bf(rowp[k0 + khalf * 8 + e] * scale);
#pragma unroll
  for (int e = 0; e < 8; ++e) v[8 + e] = f2bf(rowp[k0 + 16 + khalf * 8 + e] * scale);
  return v;
}

// ---------------------------------------------------------------------------
// Tensor Data Mover: 2D bf16 tile global->LDS (ISA 08_async_tensor D# layout).
// This toolchain exposes the 6-arg builtin:
//   (uint32x4 g0, int32x8 g1, int32x4 g2, int32x4 g3, int32x8 g?, i32 cpol)
// ---------------------------------------------------------------------------
#if __has_builtin(__builtin_amdgcn_tensor_load_to_lds)
#define HAVE_TDM 1
typedef __attribute__((ext_vector_type(4))) unsigned u32x4;
typedef __attribute__((ext_vector_type(8))) int      i32x8;
typedef __attribute__((ext_vector_type(4))) int      i32x4;

__device__ __forceinline__ void tdm_load_2d_bf16(const unsigned short* lds_dst,
                                                 const unsigned short* gsrc,
                                                 unsigned tile_w, unsigned tile_h,
                                                 unsigned long long stride_elems,
                                                 unsigned tensor_w, unsigned tensor_h)
{
  unsigned lds_off = (unsigned)(unsigned long long)lds_dst;      // LDS offset = addr[31:0]
  unsigned long long ga = (unsigned long long)gsrc;
  u32x4 g0;
  g0[0] = 1u;                                                    // count=1, user mode
  g0[1] = lds_off;                                               // lds_addr [63:32]
  g0[2] = (unsigned)(ga & 0xFFFFFFFFull);                        // global_addr [95:64]
  g0[3] = (unsigned)((ga >> 32) & 0x1FFFFFFull) | (2u << 30);    // addr hi + type=2
  i32x8 g1;
  g1[0] = 0x10000;                                               // data_size=1 -> 2 bytes
  g1[1] = (int)((tensor_w & 0xFFFFu) << 16);                     // tensor_dim0 [79:48]
  g1[2] = (int)((tensor_w >> 16) | ((tensor_h & 0xFFFFu) << 16));// + tensor_dim1 [111:80]
  g1[3] = (int)((tensor_h >> 16) | (tile_w << 16));              // + tile_dim0 [127:112]
  g1[4] = (int)(tile_h & 0xFFFFu);                               // tile_dim1; tile_dim2=0
  g1[5] = (int)(stride_elems & 0xFFFFFFFFull);                   // dim0_stride [207:160]
  g1[6] = (int)((stride_elems >> 32) & 0xFFFFull);               // dim1_stride=0
  g1[7] = 0;
  i32x4 z4 = {0, 0, 0, 0};
  i32x8 z8 = {0, 0, 0, 0, 0, 0, 0, 0};
  __builtin_amdgcn_tensor_load_to_lds(g0, g1, z4, z4, z8, 0);
}
#else
#define HAVE_TDM 0
#endif

// ---------------------------------------------------------------------------
// Generic GEMM: C(MxN) = A(MxK) * B, bf16 WMMA with fp32 accumulation.
// B_NK=true : B stored N x K row-major (C = A * B^T); false: B stored K x N.
// STORE: 0 = fp32 C, 1 = bf16 C, 2 = bf16 C transposed ([n][m]).
// 128 threads = 4 waves; workgroup tile 64(M) x 64(N); K step 32.
// ---------------------------------------------------------------------------
template<bool B_NK, int STORE>
__global__ __launch_bounds__(128)
void gemm_bf16_kernel(const float* __restrict__ A, const float* __restrict__ B,
                      void* __restrict__ Cv, int M, int N, int K,
                      long long sA, long long sB, long long sC)
{
  A += (long long)blockIdx.z * sA;
  B += (long long)blockIdx.z * sB;
  const int m0 = blockIdx.y * 64, n0 = blockIdx.x * 64;
  __shared__ __align__(16) unsigned short lA[64][32];   // [m][k]
  __shared__ __align__(16) unsigned short lB[32][64];   // [k][n]
  const int tid = threadIdx.x, w = tid >> 5, lane = tid & 31;
  v8f acc[4];
#pragma unroll
  for (int nt = 0; nt < 4; ++nt) acc[nt] = zero8();

  for (int kk = 0; kk < K; kk += 32){
    { int mr = tid >> 1, ks = (tid & 1) * 16;
      const float* src = A + (long long)(m0 + mr) * K + kk + ks;
#pragma unroll
      for (int j = 0; j < 16; ++j) lA[mr][ks + j] = f2bf(src[j]);
      if (kk + 32 < K) __builtin_prefetch(src + 32, 0, 1); }
    if (B_NK){
      int nr = tid >> 1, ks = (tid & 1) * 16;
      const float* src = B + (long long)(n0 + nr) * K + kk + ks;
#pragma unroll
      for (int j = 0; j < 16; ++j) lB[ks + j][nr] = f2bf(src[j]);
      if (kk + 32 < K) __builtin_prefetch(src + 32, 0, 1);
    } else {
      int kr = tid >> 1, ns = (tid & 1) * 32;
      const float* src = B + (long long)(kk + kr) * N + n0 + ns;
#pragma unroll
      for (int j = 0; j < 32; ++j) lB[kr][ns + j] = f2bf(src[j]);
      if (kk + 32 < K) __builtin_prefetch(src + 32LL * N, 0, 1);
    }
    __syncthreads();
    v16u16 a = ld_afrag(&lA[0][0], 32, w * 16 + (lane & 15), 0, lane >> 4);
#pragma unroll
    for (int nt = 0; nt < 4; ++nt){
      v16u16 b = ld_bfrag(&lB[0][0], 64, lane, nt * 16);
      acc[nt] = wmma_bf16(a, b, acc[nt]);
    }
    __syncthreads();
  }
  const int half = lane >> 4;
#pragma unroll
  for (int nt = 0; nt < 4; ++nt)
#pragma unroll
    for (int r = 0; r < 8; ++r){
      const int row = m0 + w * 16 + r + 8 * half;
      const int col = n0 + nt * 16 + (lane & 15);
      if (STORE == 0){
        ((float*)Cv)[(long long)blockIdx.z * sC + (long long)row * N + col] = acc[nt][r];
      } else if (STORE == 1){
        ((unsigned short*)Cv)[(long long)blockIdx.z * sC + (long long)row * N + col] =
            f2bf(acc[nt][r]);
      } else {
        ((unsigned short*)Cv)[(long long)blockIdx.z * sC + (long long)col * M + row] =
            f2bf(acc[nt][r]);
      }
    }
}

// ---------------------------------------------------------------------------
// RMSNorm / LayerNorm / RoPE / k_pe transpose
// ---------------------------------------------------------------------------
__global__ __launch_bounds__(256)
void rmsnorm_kernel(const float* __restrict__ in, int ldin,
                    const float* __restrict__ w,
                    float* __restrict__ out, int ldout, int n)
{
  const int row = blockIdx.x, tid = threadIdx.x;
  __shared__ float red[256];
  const float* ip = in + (long long)row * ldin;
  float ss = 0.f;
  for (int i = tid; i < n; i += 256){ float v = ip[i]; ss += v * v; }
  red[tid] = ss; __syncthreads();
  for (int s = 128; s > 0; s >>= 1){ if (tid < s) red[tid] += red[tid + s]; __syncthreads(); }
  float scale = rsqrtf(red[0] / n + 1e-6f);
  float* op = out + (long long)row * ldout;
  for (int i = tid; i < n; i += 256) op[i] = ip[i] * scale * w[i];
}

__global__ __launch_bounds__(128)
void layernorm_kernel(const float* __restrict__ in, const float* __restrict__ w,
                      const float* __restrict__ b, float* __restrict__ out, int n)
{
  const int row = blockIdx.x, tid = threadIdx.x;
  __shared__ float r1[128], r2[128];
  const float* ip = in + (long long)row * n;
  float v = (tid < n) ? ip[tid] : 0.f;
  r1[tid] = v; r2[tid] = v * v; __syncthreads();
  for (int s = 64; s > 0; s >>= 1){
    if (tid < s){ r1[tid] += r1[tid + s]; r2[tid] += r2[tid + s]; }
    __syncthreads();
  }
  float mu = r1[0] / n, var = r2[0] / n - mu * mu;
  float inv = rsqrtf(var + 1e-5f);
  if (tid < n) out[(long long)row * n + tid] = (v - mu) * inv * w[tid] + b[tid];
}

// traditional RoPE (interleaved even/odd pairs), in place
__global__ void rope_kernel(float* p, int ld, int hstride, int d)
{
  const int l = blockIdx.x, h = blockIdx.y, j = threadIdx.x;
  if (j >= d / 2) return;
  float* base = p + (long long)l * ld + h * hstride;
  float inv = __expf(-(2.f * j / (float)d) * __logf(10000.f));
  float ang = (float)l * inv;
  float s, c; __sincosf(ang, &s, &c);
  float x1 = base[2 * j], x2 = base[2 * j + 1];
  base[2 * j]     = x1 * c - x2 * s;
  base[2 * j + 1] = x1 * s + x2 * c;
}

// roped k_pe (fp32, inside ckv) -> bf16 transposed [d][s]
__global__ void kpe_transpose_kernel(const float* __restrict__ ckv,
                                     unsigned short* __restrict__ kpeTg)
{
  const int l = blockIdx.x, d = threadIdx.x;     // 64 threads
  kpeTg[(long long)d * L_ + l] =
      f2bf(ckv[(long long)l * (KVLORA_ + ROPED_) + KVLORA_ + d]);
}

// ---------------------------------------------------------------------------
// Lightning indexer: iscores[l][s] = sum_h relu(iq[l,h,:].ik[s,:]) * iw[l,h]
// ---------------------------------------------------------------------------
__global__ __launch_bounds__(256)
void indexer_kernel(const float* __restrict__ iq, const float* __restrict__ ik,
                    const float* __restrict__ iwraw, float* __restrict__ isc)
{
  const int l0 = blockIdx.y * 16, s0 = blockIdx.x * 128;
  const int tid = threadIdx.x, w = tid >> 5, lane = tid & 31;
  const int half = lane >> 4, col16 = lane & 15;
  const float NEG = -3.0e38f;
  if (s0 > l0 + 15){                         // fully above diagonal
#pragma unroll
    for (int r = 0; r < 8; ++r)
      isc[(long long)(l0 + r + 8 * half) * L_ + s0 + w * 16 + col16] = NEG;
    return;
  }
  __shared__ __align__(16) unsigned short ikT[128][128];  // [d][s]
  __shared__ __align__(16) unsigned short lq[16][128];    // [l][d]
  __shared__ float iwt[16][64];
  for (int idx = tid; idx < 128 * 128; idx += 256){
    int s = idx >> 7, d = idx & 127;
    ikT[d][s] = f2bf(ik[(long long)(s0 + s) * IHD_ + d]);
  }
  const float wfac = 0.011048543456039805f;  // 64^-0.5 * 128^-0.5
  for (int idx = tid; idx < 16 * 64; idx += 256){
    int lr = idx >> 6, hh = idx & 63;
    iwt[lr][hh] = iwraw[(long long)(l0 + lr) * IH_ + hh] * wfac;
  }
  v8f acc = zero8();
  for (int h = 0; h < IH_; ++h){
    __syncthreads();
    for (int idx = tid; idx < 16 * 128; idx += 256){
      int lr = idx >> 7, d = idx & 127;
      lq[lr][d] = f2bf(iq[(long long)(l0 + lr) * (IH_ * IHD_) + h * IHD_ + d]);
    }
    __syncthreads();
    v8f sc = zero8();
#pragma unroll
    for (int kc = 0; kc < 4; ++kc){
      v16u16 a = ld_afrag(&lq[0][0], 128, col16, kc * 32, half);
      v16u16 b = ld_bfrag(&ikT[0][0], 128, kc * 32 + lane, w * 16);
      sc = wmma_bf16(a, b, sc);
    }
#pragma unroll
    for (int r = 0; r < 8; ++r)
      acc[r] += fmaxf(sc[r], 0.f) * iwt[r + 8 * half][h];
  }
#pragma unroll
  for (int r = 0; r < 8; ++r){
    int row = l0 + r + 8 * half, col = s0 + w * 16 + col16;
    isc[(long long)row * L_ + col] = (col <= row) ? acc[r] : NEG;
  }
}

// ---------------------------------------------------------------------------
// Per-row top-256 via 4-pass radix select on monotonic uint keys.
// ---------------------------------------------------------------------------
__device__ __forceinline__ unsigned fkey(float f){
  unsigned u = __float_as_uint(f);
  return u ^ ((u >> 31) ? 0xFFFFFFFFu : 0x80000000u);
}

__global__ __launch_bounds__(256)
void topk_kernel(const float* __restrict__ isc, unsigned char* __restrict__ mask)
{
  const int l = blockIdx.x, tid = threadIdx.x;
  unsigned char* mrow = mask + (long long)l * L_;
  if (l < TOPK_){
    for (int s = tid; s < L_; s += 256) mrow[s] = (s <= l) ? 1 : 0;
    return;
  }
  const float* srow = isc + (long long)l * L_;
  __shared__ unsigned hist[256];
  __shared__ unsigned sh_pref, sh_rem, sh_cnt;
  unsigned prefix = 0, remaining = TOPK_;
  for (int pass = 0; pass < 4; ++pass){
    const int shift = 24 - pass * 8;
    hist[tid] = 0;
    __syncthreads();
    for (int s = tid; s <= l; s += 256){
      unsigned k = fkey(srow[s]);
      bool match = (pass == 0) || ((k >> (shift + 8)) == prefix);
      if (match) atomicAdd(&hist[(k >> shift) & 255u], 1u);
    }
    __syncthreads();
    if (tid == 0){
      unsigned rem = remaining; int b = 255;
      while (b > 0 && hist[b] < rem){ rem -= hist[b]; --b; }
      sh_pref = (prefix << 8) | (unsigned)b;
      sh_rem = rem;
    }
    __syncthreads();
    prefix = sh_pref; remaining = sh_rem;
    __syncthreads();
  }
  if (tid == 0) sh_cnt = 0;
  __syncthreads();
  const unsigned T = prefix;
  for (int s = tid; s < L_; s += 256){
    bool sel = false;
    if (s <= l){
      unsigned k = fkey(srow[s]);
      if (k > T) sel = true;
      else if (k == T){
        unsigned r = atomicAdd(&sh_cnt, 1u);
        sel = (r < remaining);
      }
    }
    mrow[s] = sel ? 1 : 0;
  }
}

// ---------------------------------------------------------------------------
// Sparse flash attention. Grid: (L/128, H). 8 waves, 16 query rows per wave.
// K (pre-transposed), k_pe (pre-transposed) and V are bf16 in global memory;
// per 32-key chunk the three tiles are staged with the TDM (fallback: manual
// LDS copies). 6 WMMAs per 16x16 score tile; online softmax; P re-laid out
// through per-wave LDS; 8 WMMAs for P.V into a 16x128 fp32 accumulator.
// ---------------------------------------------------------------------------
__global__ __launch_bounds__(256)
void attn_kernel(const float* __restrict__ q, const unsigned short* __restrict__ kbufT,
                 const unsigned short* __restrict__ vbuf,
                 const unsigned short* __restrict__ kpeTg,
                 const unsigned char* __restrict__ mask, float* __restrict__ attn)
{
  const int h = blockIdx.y;
  const int tid = threadIdx.x, w = tid >> 5, lane = tid & 31;
  const int half = lane >> 4, col16 = lane & 15;
  const int l0w = blockIdx.x * 128 + w * 16;
  const float SCALE = 0.07216878364870323f;  // 192^-0.5
  const float NEG = -3.0e38f, NEGT = -1.0e30f;

  __shared__ __align__(16) unsigned short kT[128][32];    // [d][s]
  __shared__ __align__(16) unsigned short kpeT[64][32];   // [d][s]
  __shared__ __align__(16) unsigned short lv[32][128];    // [s][vd]
  __shared__ __align__(16) unsigned short pbuf[8][16][32];

  const float* qrow = q + (long long)(l0w + col16) * (H_ * QHD_) + h * QHD_;
  v16u16 qn[4], qp[2];
#pragma unroll
  for (int kc = 0; kc < 4; ++kc) qn[kc] = ld_afrag_global(qrow, kc * 32, half, SCALE);
#pragma unroll
  for (int kc = 0; kc < 2; ++kc) qp[kc] = ld_afrag_global(qrow + NOPED_, kc * 32, half, SCALE);

  v8f o[8];
#pragma unroll
  for (int nt = 0; nt < 8; ++nt) o[nt] = zero8();
  float rmax[8], rsum[8];
#pragma unroll
  for (int r = 0; r < 8; ++r){ rmax[r] = NEG; rsum[r] = 0.f; }

  const int smax = blockIdx.x * 128 + 127;
  for (int s0 = 0; s0 <= smax; s0 += 32){
#if HAVE_TDM
    if (w == 0){
      // K^T tile: 128 rows (d) x 32 cols (s), row stride L_
      tdm_load_2d_bf16(&kT[0][0],  kbufT + (long long)h * NOPED_ * L_ + s0,
                       32, 128, L_, L_, NOPED_);
      // k_pe^T tile: 64 x 32, row stride L_
      tdm_load_2d_bf16(&kpeT[0][0], kpeTg + s0, 32, 64, L_, L_, 64);
      // V tile: 32 rows (s) x 128 cols (vd), row stride VD_
      tdm_load_2d_bf16(&lv[0][0],  vbuf + ((long long)h * L_ + s0) * VD_,
                       128, 32, VD_, VD_, L_);
      __builtin_amdgcn_s_wait_tensorcnt(0);
    }
#else
    for (int idx = tid; idx < 128 * 32; idx += 256){
      int d = idx >> 5, sl = idx & 31;
      kT[d][sl] = kbufT[((long long)h * NOPED_ + d) * L_ + s0 + sl];
    }
    for (int idx = tid; idx < 64 * 32; idx += 256){
      int d = idx >> 5, sl = idx & 31;
      kpeT[d][sl] = kpeTg[(long long)d * L_ + s0 + sl];
    }
    for (int idx = tid; idx < 32 * 128; idx += 256){
      int sl = idx >> 7, d = idx & 127;
      lv[sl][d] = vbuf[((long long)h * L_ + s0 + sl) * VD_ + d];
    }
#endif
    __syncthreads();

    v8f sc[2];
#pragma unroll
    for (int ss = 0; ss < 2; ++ss){
      sc[ss] = zero8();
#pragma unroll
      for (int kc = 0; kc < 4; ++kc)
        sc[ss] = wmma_bf16(qn[kc], ld_bfrag(&kT[0][0], 32, kc * 32 + lane, ss * 16), sc[ss]);
#pragma unroll
      for (int kc = 0; kc < 2; ++kc)
        sc[ss] = wmma_bf16(qp[kc], ld_bfrag(&kpeT[0][0], 32, kc * 32 + lane, ss * 16), sc[ss]);
#pragma unroll
      for (int r = 0; r < 8; ++r){
        int row = l0w + r + 8 * half;
        int col = s0 + ss * 16 + col16;
        if (!mask[(long long)row * L_ + col]) sc[ss][r] = NEG;
      }
    }

    float pA[8], pB[8];
#pragma unroll
    for (int r = 0; r < 8; ++r){
      float mt = fmaxf(sc[0][r], sc[1][r]);
#pragma unroll
      for (int off = 1; off < 16; off <<= 1) mt = fmaxf(mt, __shfl_xor(mt, off, 32));
      float mnew = fmaxf(rmax[r], mt);
      float alpha = __expf(rmax[r] - mnew);
      float pa = (sc[0][r] > NEGT) ? __expf(sc[0][r] - mnew) : 0.f;
      float pb = (sc[1][r] > NEGT) ? __expf(sc[1][r] - mnew) : 0.f;
      float ps = pa + pb;
#pragma unroll
      for (int off = 1; off < 16; off <<= 1) ps += __shfl_xor(ps, off, 32);
      rsum[r] = rsum[r] * alpha + ps;
      rmax[r] = mnew;
#pragma unroll
      for (int nt = 0; nt < 8; ++nt) o[nt][r] *= alpha;
      pA[r] = pa; pB[r] = pb;
    }

#pragma unroll
    for (int r = 0; r < 8; ++r){
      pbuf[w][r + 8 * half][col16]      = f2bf(pA[r]);
      pbuf[w][r + 8 * half][16 + col16] = f2bf(pB[r]);
    }
    __syncthreads();
    v16u16 pf = ld_afrag(&pbuf[w][0][0], 32, col16, 0, half);
#pragma unroll
    for (int nt = 0; nt < 8; ++nt)
      o[nt] = wmma_bf16(pf, ld_bfrag(&lv[0][0], 128, lane, nt * 16), o[nt]);
    __syncthreads();
  }

#pragma unroll
  for (int nt = 0; nt < 8; ++nt)
#pragma unroll
    for (int r = 0; r < 8; ++r){
      int row = l0w + r + 8 * half;
      float denom = rsum[r];
      float val = (denom > 0.f) ? (o[nt][r] / denom) : 0.f;
      attn[(long long)row * (H_ * VD_) + h * VD_ + nt * 16 + col16] = val;
    }
}

// ---------------------------------------------------------------------------
extern "C" void kernel_launch(void* const* d_in, const int* in_sizes, int n_in,
                              void* d_out, int out_size, void* d_ws, size_t ws_size,
                              hipStream_t stream)
{
  (void)in_sizes; (void)n_in; (void)out_size; (void)ws_size;
  const float* x         = (const float*)d_in[0];
  /* d_in[1] = bool causal mask (implicit here) */
  const float* W_qa      = (const float*)d_in[2];
  const float* qa_ln_w   = (const float*)d_in[3];
  const float* W_qb      = (const float*)d_in[4];
  const float* W_kva     = (const float*)d_in[5];
  const float* kva_ln_w  = (const float*)d_in[6];
  const float* W_eq      = (const float*)d_in[7];
  const float* W_uo      = (const float*)d_in[8];
  const float* W_o       = (const float*)d_in[9];
  const float* idx_wqb   = (const float*)d_in[10];
  const float* idx_wk    = (const float*)d_in[11];
  const float* idx_kln_w = (const float*)d_in[12];
  const float* idx_kln_b = (const float*)d_in[13];
  const float* idx_wproj = (const float*)d_in[14];
  float* out = (float*)d_out;

  float* ws = (float*)d_ws;
  size_t off = 0;
  float* qr    = ws + off; off += (size_t)L_ * QLORA_;
  float* q     = ws + off; off += (size_t)L_ * H_ * QHD_;
  float* ckv   = ws + off; off += (size_t)L_ * (KVLORA_ + ROPED_);
  float* kvlat = ws + off; off += (size_t)L_ * KVLORA_;
  float* iq    = ws + off; off += (size_t)L_ * IH_ * IHD_;
  float* ik    = ws + off; off += (size_t)L_ * IHD_;
  float* iw    = ws + off; off += (size_t)L_ * IH_;
  float* isc   = ws + off; off += (size_t)L_ * L_;
  float* attn  = ws + off; off += (size_t)L_ * H_ * VD_;
  unsigned char*  maskbuf = (unsigned char*)(ws + off);
  unsigned short* kbufT   = (unsigned short*)(maskbuf + (size_t)L_ * L_);     // [h][d][s]
  unsigned short* vbuf    = kbufT + (size_t)H_ * NOPED_ * L_;                 // [h][s][vd]
  unsigned short* kpeTg   = vbuf + (size_t)H_ * L_ * VD_;                     // [d][s]

  // --- projections from x ---
  gemm_bf16_kernel<true, 0><<<dim3(QLORA_/64, L_/64, 1), 128, 0, stream>>>(
      x, W_qa, qr, L_, QLORA_, D_, 0, 0, 0);
  gemm_bf16_kernel<true, 0><<<dim3((KVLORA_+ROPED_)/64, L_/64, 1), 128, 0, stream>>>(
      x, W_kva, ckv, L_, KVLORA_+ROPED_, D_, 0, 0, 0);
  gemm_bf16_kernel<true, 0><<<dim3(IHD_/64, L_/64, 1), 128, 0, stream>>>(
      x, idx_wk, ik, L_, IHD_, D_, 0, 0, 0);
  gemm_bf16_kernel<true, 0><<<dim3(IH_/64, L_/64, 1), 128, 0, stream>>>(
      x, idx_wproj, iw, L_, IH_, D_, 0, 0, 0);

  // --- norms + RoPE on latents ---
  rmsnorm_kernel<<<L_, 256, 0, stream>>>(qr, QLORA_, qa_ln_w, qr, QLORA_, QLORA_);
  rmsnorm_kernel<<<L_, 256, 0, stream>>>(ckv, KVLORA_+ROPED_, kva_ln_w, kvlat, KVLORA_, KVLORA_);
  layernorm_kernel<<<L_, 128, 0, stream>>>(ik, idx_kln_w, idx_kln_b, ik, IHD_);
  rope_kernel<<<dim3(L_, 1), 32, 0, stream>>>(ckv + KVLORA_, KVLORA_+ROPED_, 0, ROPED_);
  rope_kernel<<<dim3(L_, 1), 32, 0, stream>>>(ik, IHD_, 0, ROPED_);
  kpe_transpose_kernel<<<L_, ROPED_, 0, stream>>>(ckv, kpeTg);

  // --- q / index-q up-projections + RoPE ---
  gemm_bf16_kernel<true, 0><<<dim3(H_*QHD_/64, L_/64, 1), 128, 0, stream>>>(
      qr, W_qb, q, L_, H_*QHD_, QLORA_, 0, 0, 0);
  gemm_bf16_kernel<true, 0><<<dim3(IH_*IHD_/64, L_/64, 1), 128, 0, stream>>>(
      qr, idx_wqb, iq, L_, IH_*IHD_, QLORA_, 0, 0, 0);
  rope_kernel<<<dim3(L_, H_), 32, 0, stream>>>(q + NOPED_, H_*QHD_, QHD_, ROPED_);
  rope_kernel<<<dim3(L_, IH_), 32, 0, stream>>>(iq, IH_*IHD_, IHD_, ROPED_);

  // --- per-head K / V expansion (bf16 outputs; K stored transposed) ---
  gemm_bf16_kernel<false, 2><<<dim3(NOPED_/64, L_/64, H_), 128, 0, stream>>>(
      kvlat, W_eq, kbufT, L_, NOPED_, KVLORA_,
      0, (long long)KVLORA_*NOPED_, (long long)NOPED_*L_);
  gemm_bf16_kernel<true, 1><<<dim3(VD_/64, L_/64, H_), 128, 0, stream>>>(
      kvlat, W_uo, vbuf, L_, VD_, KVLORA_,
      0, (long long)VD_*KVLORA_, (long long)L_*VD_);

  // --- lightning indexer + top-k sparse mask ---
  indexer_kernel<<<dim3(L_/128, L_/16, 1), 256, 0, stream>>>(iq, ik, iw, isc);
  topk_kernel<<<L_, 256, 0, stream>>>(isc, maskbuf);

  // --- sparse flash attention + output projection ---
  attn_kernel<<<dim3(L_/128, H_, 1), 256, 0, stream>>>(q, kbufT, vbuf, kpeTg, maskbuf, attn);
  gemm_bf16_kernel<true, 0><<<dim3(D_/64, L_/64, 1), 128, 0, stream>>>(
      attn, W_o, out, L_, D_, H_*VD_, 0, 0, 0);
}
